// GaussianNNHeightmapEncoder_26852135535197
// MI455X (gfx1250) — matrix operand order repaired and checked by reading, not compiled, not run on hardware
//
#include <hip/hip_runtime.h>

typedef __attribute__((ext_vector_type(16))) _Float16 v16h;
typedef __attribute__((ext_vector_type(8)))  _Float16 v8h;
typedef __attribute__((ext_vector_type(4)))  _Float16 v4h;
typedef __attribute__((ext_vector_type(8)))  float    v8f;

#define NEG_SLOPE 0.01f

union HFrag { v16h v; _Float16 e[16]; };

__device__ __forceinline__ float lrelu(float x) { return x > 0.f ? x : NEG_SLOPE * x; }

__device__ __forceinline__ v8f vzero8() {
  v8f z;
#pragma unroll
  for (int i = 0; i < 8; ++i) z[i] = 0.f;
  return z;
}

// A fragment (16x32 f16) from row-major [.,64] LDS: one contiguous 16-half run -> 2x ds_load_b128
__device__ __forceinline__ v16h lds_a_frag(const _Float16* M, int m0, int k0, int lane) {
  HFrag f;
  const _Float16* row = M + (m0 + (lane & 15)) * 64 + k0 + ((lane >> 4) << 4);
#pragma unroll
  for (int r = 0; r < 8; ++r) { f.e[r] = row[r]; f.e[8 + r] = row[8 + r]; }
  return f.v;
}

// A fragment where M rows are flattened [4096] per-batch buffers (compress stage).
__device__ __forceinline__ v16h lds_a_frag_flat(const _Float16* HBbase, int k0, int lane) {
  HFrag f;
  const _Float16* hb = HBbase + (size_t)((lane & 15) & 3) * 4096 + k0 + ((lane >> 4) << 4);
#pragma unroll
  for (int r = 0; r < 8; ++r) { f.e[r] = hb[r]; f.e[8 + r] = hb[8 + r]; }
  return f.v;
}

// B fragment from global f32 weight [N][K] row-major: 16 consecutive floats/lane + cvt
__device__ __forceinline__ v16h glb_b_frag_f32(const float* __restrict__ W, int ld,
                                               int n0, int k0, int lane) {
  HFrag f;
  const float* row = W + (size_t)(n0 + (lane & 15)) * ld + k0 + ((lane >> 4) << 4);
#pragma unroll
  for (int r = 0; r < 16; ++r) f.e[r] = (_Float16)row[r];
  return f.v;
}

// B fragment from pre-converted global f16 weight [N][K]: one 32B load, zero VALU
__device__ __forceinline__ v16h glb_b_frag_f16(const _Float16* __restrict__ W, int ld,
                                               int n0, int k0, int lane) {
  return *(const v16h*)(W + (size_t)(n0 + (lane & 15)) * ld + k0 + ((lane >> 4) << 4));
}

// ---------------- templated gaussian layer (fully unrolled WMMA) ----------------
template <int CIN, int COUT>
__device__ __forceinline__ void gauss_layer(_Float16* HBw, _Float16* XPw,
                                            const _Float16* KA, const _Float16* WTs,
                                            const float* __restrict__ bias, int lane) {
  constexpr int NT  = (COUT + 15) / 16;
  constexpr int KS1 = (CIN > 32) ? 2 : 1;
  const int n15 = lane & 15;
  const int hb  = lane >> 4;

  v8f acc2[4][NT];
#pragma unroll
  for (int i = 0; i < 4; ++i)
#pragma unroll
    for (int j = 0; j < NT; ++j) acc2[i][j] = vzero8();

  float biasv[NT];
#pragma unroll
  for (int nt = 0; nt < NT; ++nt) {
    int n = nt * 16 + n15;
    biasv[nt] = (n < COUT) ? bias[n] : 0.f;
  }

#pragma unroll
  for (int phase = 0; phase < 2; ++phase) {
    // ---- GEMM1: xp rows [phase*32, +32) = h @ W^T + bias, stored swizzled ----
#pragma unroll
    for (int mt = 0; mt < 2; ++mt) {
      const int m0 = phase * 32 + mt * 16;
      v16h a[KS1];
#pragma unroll
      for (int ks = 0; ks < KS1; ++ks) a[ks] = lds_a_frag(HBw, m0, ks * 32, lane);
#pragma unroll
      for (int nt = 0; nt < NT; ++nt) {
        v8f acc = vzero8();
#pragma unroll
        for (int ks = 0; ks < KS1; ++ks) {
          v16h bf = *(const v16h*)(WTs + (ks * 4 + nt) * 512 + lane * 16);
          acc = __builtin_amdgcn_wmma_f32_16x16x32_f16(false, a[ks], false, bf,
                                                       (short)0, acc, false, false);
        }
        v8h pk;
#pragma unroll
        for (int r = 0; r < 8; ++r) pk[r] = (_Float16)(acc[r] + biasv[nt]);
        // C rows (m = mt*16 + 8*hb + r) -> swizzled B positions: one b128 store
        *(v8h*)(XPw + nt * 512 + (n15 + 16 * mt) * 16 + 8 * hb) = pk;
      }
    }
    // ---- GEMM2 accumulate: h_next += k[:, K-slice] @ xp-chunk (same-wave DS is in-order) ----
#pragma unroll
    for (int mt = 0; mt < 4; ++mt) {
      v16h a = *(const v16h*)(KA + (mt * 2 + phase) * 512 + lane * 16);
#pragma unroll
      for (int nt = 0; nt < NT; ++nt) {
        v16h bf = *(const v16h*)(XPw + nt * 512 + lane * 16);
        acc2[mt][nt] = __builtin_amdgcn_wmma_f32_16x16x32_f16(false, a, false, bf,
                                                              (short)0, acc2[mt][nt],
                                                              false, false);
      }
    }
  }

  // writeback: leaky-relu, f16, zero-pad cols [NT*16, 32) for next layer's clean K=32 read
#pragma unroll
  for (int mt = 0; mt < 4; ++mt)
#pragma unroll
    for (int nt = 0; nt < NT; ++nt) {
      const int nn = nt * 16 + n15;
#pragma unroll
      for (int r = 0; r < 8; ++r)
        HBw[(mt * 16 + 8 * hb + r) * 64 + nn] = (_Float16)lrelu(acc2[mt][nt][r]);
    }
  if (NT * 16 < 32) {
    for (int c = NT * 16 + lane; c < 32; c += 32)
      for (int row = 0; row < 64; ++row) HBw[row * 64 + c] = (_Float16)0.f;
  }
}

// Stage W^T (zero-padded) into swizzled B-frag blocks and the L1-normalized gaussian
// kernel into swizzled A-frag blocks. Only 64 expf per layer (distance LUT).
__device__ __forceinline__ void stage_layer(const float* __restrict__ W,
                                            const float* __restrict__ S,
                                            int cin, int cout,
                                            _Float16* WTs, _Float16* KA,
                                            float* g, float* rsc, int tid) {
  __syncthreads();  // previous users of WTs/KA/XP region are done
  if (tid < 64) {
    float sg  = S[0];
    float inv = -1.f / (2.f * sg * sg);
    float d   = (float)tid;
    g[tid] = __expf(d * d * inv);
  }
  // W^T swizzled: element (k,n) -> block (k/32)*4 + n/16, lane (n&15)+16*((k&31)>=16), pos k&15
  for (int i = tid; i < 4096; i += 128) {
    int k = i >> 6, n = i & 63;
    float v = (k < cin && n < cout) ? W[n * cin + k] : 0.f;
    int kk = k & 31;
    int addr = ((k >> 5) * 4 + (n >> 4)) * 512 + (((kk >> 4) << 4) + (n & 15)) * 16 + (kk & 15);
    WTs[addr] = (_Float16)v;
  }
  __syncthreads();
  if (tid < 64) {
    float sum = 0.f;
    for (int j = 0; j < 64; ++j) sum += g[tid >= j ? tid - j : j - tid];
    rsc[tid] = 1.f / fmaxf(sum, 1e-12f);
  }
  __syncthreads();
  // gaussian kernel swizzled A-frags: element (i=m, j=k):
  //   block = (i>>4)*2 + (j>>5); kk=j&31; hb=(kk>>4)&1; pos=kk&15
  for (int idx = tid; idx < 4096; idx += 128) {
    int i = idx >> 6, j = idx & 63;
    int d = i >= j ? i - j : j - i;
    float v = g[d] * rsc[i];
    int kk = j & 31;
    int hb2 = (kk >> 4) & 1;
    int pos = kk & 15;
    int addr = ((i >> 4) * 2 + (j >> 5)) * 512 + ((i & 15) + 16 * hb2) * 16 + pos;
    KA[addr] = (_Float16)v;
  }
  __syncthreads();
}

// Pre-convert wm0 (80x4096 f32) to f16 in workspace: removes ~330K cvts per main block.
__global__ __launch_bounds__(256)
void wm0_to_f16_kernel(const float* __restrict__ wm0, _Float16* __restrict__ wm0h) {
  const int i = (blockIdx.x * 256 + threadIdx.x) * 4;   // 80*4096 = 327680 elems
  float4 v = *(const float4*)(wm0 + i);
  v4h p;
  p[0] = (_Float16)v.x; p[1] = (_Float16)v.y;
  p[2] = (_Float16)v.z; p[3] = (_Float16)v.w;
  *(v4h*)(wm0h + i) = p;
}

template <bool WM0H>
__global__ __launch_bounds__(128)
void GaussianNNHeightmapEncoder_kernel(
    const float* __restrict__ x,
    const float* __restrict__ w0, const float* __restrict__ b0, const float* __restrict__ s0,
    const float* __restrict__ w1, const float* __restrict__ b1, const float* __restrict__ s1,
    const float* __restrict__ w2, const float* __restrict__ b2, const float* __restrict__ s2,
    const float* __restrict__ w3, const float* __restrict__ b3, const float* __restrict__ s3,
    const float* __restrict__ wm0, const _Float16* __restrict__ wm0h,
    const float* __restrict__ bm0,
    const float* __restrict__ wm1, const float* __restrict__ bm1,
    float* __restrict__ out, int B)
{
  // exactly 64 KB of LDS
  __shared__ __align__(32) _Float16 HB[4][4096];  // per-wave h, row-major [64][64]
  __shared__ __align__(32) _Float16 XP[4][2048];  // per-wave swizzled xp; reused: g/rsc, YB
  __shared__ __align__(32) _Float16 KA[4096];     // gaussian kernel, swizzled A-frag blocks
  __shared__ __align__(32) _Float16 WT[4096];     // W^T, swizzled B-frag blocks

  const int tid  = threadIdx.x;
  const int wave = tid >> 5;
  const int lane = tid & 31;
  int b = blockIdx.x * 4 + wave;
  if (b >= B) b = B - 1;             // duplicate work; output writes are guarded

  _Float16* HBw = HB[wave];
  _Float16* XPw = XP[wave];
  float* g   = (float*)&XP[0][0];    // 64 floats, staging lifetime only
  float* rsc = g + 64;               // 64 floats

  // load x[b] -> HB as f16 (float4 loads, b64 LDS stores)
  {
    const float4* xb4 = (const float4*)(x + (size_t)b * 4096);
    for (int i = lane; i < 1024; i += 32) {
      float4 v = xb4[i];
      v4h p;
      p[0] = (_Float16)v.x; p[1] = (_Float16)v.y;
      p[2] = (_Float16)v.z; p[3] = (_Float16)v.w;
      *(v4h*)(HBw + i * 4) = p;
    }
  }

  stage_layer(w0, s0, 64, 8,  WT, KA, g, rsc, tid);
  gauss_layer<64, 8 >(HBw, XPw, KA, WT, b0, lane);
  stage_layer(w1, s1, 8,  16, WT, KA, g, rsc, tid);
  gauss_layer<8,  16>(HBw, XPw, KA, WT, b1, lane);
  stage_layer(w2, s2, 16, 32, WT, KA, g, rsc, tid);
  gauss_layer<16, 32>(HBw, XPw, KA, WT, b2, lane);
  stage_layer(w3, s3, 32, 64, WT, KA, g, rsc, tid);
  gauss_layer<32, 64>(HBw, XPw, KA, WT, b3, lane);

  __syncthreads();  // cross-wave HB reads in compress

  // ---- Compress 1 via WMMA: batches in M (4 valid rows), K=4096, N=80 (5 tiles) ----
  float* YB = (float*)&XP[0][0];     // [4][80] activations
  {
    const int n15 = lane & 15;
    const int hi  = lane >> 4;
    for (int t = wave; t < 5; t += 4) {
      v8f acc = vzero8();
      for (int ks = 0; ks < 128; ++ks) {
        v16h a = lds_a_frag_flat(&HB[0][0], ks * 32, lane);
        v16h bf;
        if (WM0H) bf = glb_b_frag_f16(wm0h, 4096, t * 16, ks * 32, lane);
        else      bf = glb_b_frag_f32(wm0,  4096, t * 16, ks * 32, lane);
        acc = __builtin_amdgcn_wmma_f32_16x16x32_f16(false, a, false, bf,
                                                     (short)0, acc, false, false);
      }
      const int n     = t * 16 + n15;
      const float bia = bm0[n];
#pragma unroll
      for (int r = 0; r < 8; ++r) {
        const int m = r + (hi ? 8 : 0);
        if (m < 4) YB[m * 80 + n] = lrelu(acc[r] + bia);
      }
    }
  }
  __syncthreads();

  // ---- Compress 2: tiny 80->60 scalar pass ----
  for (int idx = tid; idx < 4 * 60; idx += 128) {
    const int bb = idx / 60, o = idx % 60;
    const float* wr = wm1 + o * 80;
    float sum = bm1[o];
    for (int j = 0; j < 80; ++j) sum += YB[bb * 80 + j] * wr[j];
    const int gb = blockIdx.x * 4 + bb;
    if (gb < B) out[(size_t)gb * 60 + o] = lrelu(sum);
  }
}

extern "C" void kernel_launch(void* const* d_in, const int* in_sizes, int n_in,
                              void* d_out, int out_size, void* d_ws, size_t ws_size,
                              hipStream_t stream) {
  const float* x   = (const float*)d_in[0];
  const float* w0  = (const float*)d_in[1];
  const float* b0  = (const float*)d_in[2];
  const float* s0  = (const float*)d_in[3];
  const float* w1  = (const float*)d_in[4];
  const float* b1  = (const float*)d_in[5];
  const float* s1  = (const float*)d_in[6];
  const float* w2  = (const float*)d_in[7];
  const float* b2  = (const float*)d_in[8];
  const float* s2  = (const float*)d_in[9];
  const float* w3  = (const float*)d_in[10];
  const float* b3  = (const float*)d_in[11];
  const float* s3  = (const float*)d_in[12];
  const float* wm0 = (const float*)d_in[13];
  const float* bm0 = (const float*)d_in[14];
  const float* wm1 = (const float*)d_in[15];
  const float* bm1 = (const float*)d_in[16];
  float* out = (float*)d_out;

  const int B = in_sizes[0] / 4096;
  if (B <= 0) return;
  const int blocks = (B + 3) / 4;

  const size_t wm0h_bytes = (size_t)80 * 4096 * sizeof(_Float16);  // 655360
  if (ws_size >= wm0h_bytes) {
    _Float16* wm0h = (_Float16*)d_ws;
    wm0_to_f16_kernel<<<(80 * 4096) / (256 * 4), 256, 0, stream>>>(wm0, wm0h);
    GaussianNNHeightmapEncoder_kernel<true><<<blocks, 128, 0, stream>>>(
        x, w0, b0, s0, w1, b1, s1, w2, b2, s2, w3, b3, s3,
        wm0, wm0h, bm0, wm1, bm1, out, B);
  } else {
    GaussianNNHeightmapEncoder_kernel<false><<<blocks, 128, 0, stream>>>(
        x, w0, b0, s0, w1, b1, s1, w2, b2, s2, w3, b3, s3,
        wm0, (const _Float16*)nullptr, bm0, wm1, bm1, out, B);
  }
}
